// MyGRU_23682449670336
// MI455X (gfx1250) — matrix operand order repaired and checked by reading, not compiled, not run on hardware
//
#include <hip/hip_runtime.h>

// ---------------------------------------------------------------------------
// GRU scan, T=100, B=4096, D=U=128.  Each workgroup owns 32 batch rows and
// runs the full recurrence with weights resident in LDS (bf16).  The f32
// master state lives in REGISTERS (each element owned by one wave for all
// 100 steps); only the bf16 matmul copy goes through LDS.  Matmuls use
// V_WMMA_F32_16X16X32_BF16 with f32 accumulation.
// Round 3: amdgpu_waves_per_eu(2,2) (LDS limits us to 1 WG/WGP = 2 waves/SIMD,
// so let the allocator keep the 12-fragment B batch live), register-resident
// f32 state.
// ---------------------------------------------------------------------------

typedef __bf16 bf16;
typedef __attribute__((ext_vector_type(16))) __bf16 v16bf;
typedef __attribute__((ext_vector_type(8)))  __bf16 v8bf;
typedef __attribute__((ext_vector_type(8)))  float  v8f;

#define T_STEPS 100
#define B_TOT   4096
#define D_DIM   128
#define U_DIM   128
#define BT      32     // batch rows per workgroup
#define PW      136    // bf16 row pitch (128 + 8 pad -> conflict-free frag loads)

// LDS bytes: Wx+Wh (2*384*PW*2) + X double buf (2*BT*PW*2) + Hbf (BT*PW*2)
//          + bias (384*4) + mask double buf (2*BT*4) = 236,800
#define SMEM_BYTES (2*(384*PW*2) + 2*(BT*PW*2) + (BT*PW*2) + 384*4 + 2*BT*4)

// ---- A fragment: 16x32 bf16, rows rowBase..rowBase+15, K chunk kc*32 -------
__device__ __forceinline__ v16bf load_frag_a(const bf16* base, int rowBase,
                                             int kc, int lane) {
    int m    = lane & 15;
    int ksel = (lane >> 4) << 3;                   // 0 or 8
    const bf16* p = base + (size_t)(rowBase + m) * PW + kc * 32 + ksel;
    v8bf lo = *(const v8bf*)(p);                   // K = base .. base+7
    v8bf hi = *(const v8bf*)(p + 16);              // K = base+16 .. base+23
    v16bf a;
#pragma unroll
    for (int i = 0; i < 8; ++i) { a[i] = lo[i]; a[i + 8] = hi[i]; }
    return a;
}

// ---- B fragment: 32x16 bf16 from N-major (transposed) weights --------------
__device__ __forceinline__ v16bf load_frag_b(const bf16* wt, int n0,
                                             int kc, int lane) {
    int n  = n0 + (lane & 15);
    int kb = kc * 32 + ((lane >> 4) << 3);         // 0 or 8
    const bf16* p = wt + (size_t)n * PW + kb;
    v8bf lo = *(const v8bf*)(p);
    v8bf hi = *(const v8bf*)(p + 16);
    v16bf b;
#pragma unroll
    for (int i = 0; i < 8; ++i) { b[i] = lo[i]; b[i + 8] = hi[i]; }
    return b;
}

// stage one 128x128 f32 weight matrix into LDS, transposed (N-major), bf16
__device__ __forceinline__ void stage_w(bf16* dst, const float* __restrict__ src,
                                        int tid) {
    for (int idx = tid; idx < 128 * 128; idx += 256) {
        int n = idx & 127, k = idx >> 7;           // coalesced global reads
        dst[(size_t)n * PW + k] = (bf16)src[(size_t)k * 128 + n];
    }
}

// stage one timestep's x tile (f32 -> bf16) and mask column
__device__ __forceinline__ void stage_xt(bf16* dstX, float* dstM,
                                         const float* __restrict__ inputs,
                                         const float* __restrict__ mask,
                                         int t, int b0, int tid) {
    for (int idx = tid; idx < BT * 128; idx += 256) {
        int r = idx >> 7, d = idx & 127;
        dstX[r * PW + d] = (bf16)inputs[((size_t)t * B_TOT + b0 + r) * D_DIM + d];
    }
    if (tid < BT) dstM[tid] = mask[(size_t)(b0 + tid) * T_STEPS + t];
}

__device__ __forceinline__ float sigmoid_f(float z) {
    return 1.0f / (1.0f + __expf(-z));
}
__device__ __forceinline__ float tanh_f(float z) {
    return 1.0f - 2.0f / (__expf(2.0f * z) + 1.0f);
}

#define WMMA_BF16(A, B, C) \
    __builtin_amdgcn_wmma_f32_16x16x32_bf16(false, (A), false, (B), (short)0, (C), false, false)

extern "C" __global__ __launch_bounds__(256)
__attribute__((amdgpu_waves_per_eu(2, 2)))   // 236KB LDS -> 1 WG/WGP, 2 waves/SIMD
void gru_scan_kernel(const float* __restrict__ inputs, const float* __restrict__ state,
                     const float* __restrict__ mask,
                     const float* __restrict__ Wux, const float* __restrict__ bu,
                     const float* __restrict__ Wuh,
                     const float* __restrict__ Wrx, const float* __restrict__ br,
                     const float* __restrict__ Wrh,
                     const float* __restrict__ Wcx, const float* __restrict__ bc,
                     const float* __restrict__ Wch,
                     float* __restrict__ out) {
    extern __shared__ char smem[];
    bf16*  sWx   = (bf16*)smem;                    // [384][PW]  x-proj weights (u|r|c)
    bf16*  sWh   = sWx + 384 * PW;                 // [384][PW]  h-proj weights
    bf16*  sX    = sWh + 384 * PW;                 // [2][BT][PW] x_t double buffer
    bf16*  sHbf  = sX  + 2 * BT * PW;              // [BT][PW]   h (bf16, matmul copy)
    float* sBias = (float*)(sHbf + BT * PW);       // [384]
    float* sMask = sBias + 384;                    // [2][BT]    mask double buffer

    const int tid  = threadIdx.x;
    const int lane = tid & 31;
    const int wave = tid >> 5;
    const int b0   = blockIdx.x * BT;

    // wave -> (row tile, pair of U-column tiles); fixed for the whole scan,
    // so this wave exclusively owns h[(mt*16..+15), (jt0*16..jt0*16+31)].
    const int mt  = wave >> 2;                     // 0..1  -> rows mt*16..mt*16+15
    const int jt0 = (wave & 3) * 2;                // U col tiles jt0, jt0+1

    // ---- one-time staging: weights (bf16, transposed), bias, initial state
    stage_w(sWx + 0 * 128 * PW, Wux, tid);
    stage_w(sWx + 1 * 128 * PW, Wrx, tid);
    stage_w(sWx + 2 * 128 * PW, Wcx, tid);
    stage_w(sWh + 0 * 128 * PW, Wuh, tid);
    stage_w(sWh + 1 * 128 * PW, Wrh, tid);
    stage_w(sWh + 2 * 128 * PW, Wch, tid);
    for (int n = tid; n < 128; n += 256) {
        sBias[n]       = bu[n];
        sBias[128 + n] = br[n];
        sBias[256 + n] = bc[n];
    }
    for (int idx = tid; idx < BT * 128; idx += 256) {
        int r = idx >> 7, u = idx & 127;
        sHbf[r * PW + u] = (bf16)state[(size_t)(b0 + r) * U_DIM + u];
    }
    stage_xt(sX, sMask, inputs, mask, 0, b0, tid);   // t=0 into buffer 0
    __syncthreads();

    // time-invariant per-lane values + register-resident f32 state
    const int rowv = mt * 16 + ((lane >> 4) << 3); // row for v=0 (C/D layout)
    float bju[2], bjr[2], bjc[2], hreg[2][8];
    int   jcol[2], off_b[2];
#pragma unroll
    for (int jj = 0; jj < 2; ++jj) {
        const int j = (jt0 + jj) * 16 + (lane & 15);
        jcol[jj]  = j;
        bju[jj]   = sBias[j];
        bjr[jj]   = sBias[128 + j];
        bjc[jj]   = sBias[256 + j];
        off_b[jj] = rowv * PW + j;                 // sHbf offset, v stride PW
#pragma unroll
        for (int v = 0; v < 8; ++v)
            hreg[jj][v] = state[(size_t)(b0 + rowv + v) * U_DIM + j];
    }

    for (int t = 0; t < T_STEPS; ++t) {
        const bf16*  sXc = sX + (t & 1) * BT * PW;
        const float* sMc = sMask + (t & 1) * BT;

        // ---- GEMM phase: gates = X@Wx + H@Wh, f32 accum via bf16 WMMA -----
        v8f acc_u[2], acc_r[2], acc_cx[2], acc_ch[2];
#pragma unroll
        for (int jj = 0; jj < 2; ++jj) {
            acc_u[jj] = (v8f){};  acc_r[jj] = (v8f){};
            acc_cx[jj] = (v8f){}; acc_ch[jj] = (v8f){};
        }
        v16bf ax[4], ah[4];
#pragma unroll
        for (int kc = 0; kc < 4; ++kc) {
            ax[kc] = load_frag_a(sXc,  mt * 16, kc, lane);
            ah[kc] = load_frag_a(sHbf, mt * 16, kc, lane);
        }
#pragma unroll
        for (int kc = 0; kc < 4; ++kc) {
            // batch all 12 B-fragment loads of this K-chunk (24 ds_load_b128)
            v16bf bq[12];
#pragma unroll
            for (int jj = 0; jj < 2; ++jj) {
                const int nu = (jt0 + jj) * 16;
                bq[0  + jj] = load_frag_b(sWx, nu,       kc, lane);
                bq[2  + jj] = load_frag_b(sWx, 128 + nu, kc, lane);
                bq[4  + jj] = load_frag_b(sWx, 256 + nu, kc, lane);
                bq[6  + jj] = load_frag_b(sWh, nu,       kc, lane);
                bq[8  + jj] = load_frag_b(sWh, 128 + nu, kc, lane);
                bq[10 + jj] = load_frag_b(sWh, 256 + nu, kc, lane);
            }
            // 12 WMMAs; the two updates of each accumulator are 6 ops apart
#pragma unroll
            for (int jj = 0; jj < 2; ++jj) acc_u[jj]  = WMMA_BF16(ax[kc], bq[0  + jj], acc_u[jj]);
#pragma unroll
            for (int jj = 0; jj < 2; ++jj) acc_r[jj]  = WMMA_BF16(ax[kc], bq[2  + jj], acc_r[jj]);
#pragma unroll
            for (int jj = 0; jj < 2; ++jj) acc_cx[jj] = WMMA_BF16(ax[kc], bq[4  + jj], acc_cx[jj]);
#pragma unroll
            for (int jj = 0; jj < 2; ++jj) acc_u[jj]  = WMMA_BF16(ah[kc], bq[6  + jj], acc_u[jj]);
#pragma unroll
            for (int jj = 0; jj < 2; ++jj) acc_r[jj]  = WMMA_BF16(ah[kc], bq[8  + jj], acc_r[jj]);
#pragma unroll
            for (int jj = 0; jj < 2; ++jj) acc_ch[jj] = WMMA_BF16(ah[kc], bq[10 + jj], acc_ch[jj]);
        }
        __syncthreads();   // all Hbf/X reads done before state update

        // ---- elementwise gate math + state update (state in registers) ----
        // C/D layout: lane<16 -> N=lane, M=v ; lane>=16 -> N=lane-16, M=v+8
#pragma unroll
        for (int jj = 0; jj < 2; ++jj) {
#pragma unroll
            for (int v = 0; v < 8; ++v) {
                float h  = hreg[jj][v];
                float m  = sMc[rowv + v];
                float u  = sigmoid_f(acc_u[jj][v] + bju[jj]);
                float r  = sigmoid_f(acc_r[jj][v] + bjr[jj]);
                float c  = tanh_f(acc_cx[jj][v] + bjc[jj] + r * acc_ch[jj][v]);
                float tmp = (1.0f - u) * h + u * c;
                float h1  = tmp * m + h * (1.0f - m);
                float h2  = (1.0f - u) * h1 + u * c;
                hreg[jj][v] = h2;
                sHbf[off_b[jj] + v * PW] = (bf16)h2;
            }
        }

        // ---- stage next timestep into the other buffer (overlaps elem) ----
        if (t + 1 < T_STEPS) {
            stage_xt(sX + ((t + 1) & 1) * BT * PW, sMask + ((t + 1) & 1) * BT,
                     inputs, mask, t + 1, b0, tid);
            if (t + 2 < T_STEPS) {   // pull t+2's x tile toward L2
                const char* nx =
                    (const char*)&inputs[((size_t)(t + 2) * B_TOT + b0) * D_DIM];
                __builtin_prefetch(nx + tid * 64, 0, 1);
            }
        }
        __syncthreads();   // state update + staging visible before next GEMM
    }

    // ---- write final state straight from registers ------------------------
#pragma unroll
    for (int jj = 0; jj < 2; ++jj)
#pragma unroll
        for (int v = 0; v < 8; ++v)
            out[(size_t)(b0 + rowv + v) * U_DIM + jcol[jj]] = hreg[jj][v];
}

extern "C" void kernel_launch(void* const* d_in, const int* in_sizes, int n_in,
                              void* d_out, int out_size, void* d_ws, size_t ws_size,
                              hipStream_t stream) {
    const float* inputs = (const float*)d_in[0];
    const float* state  = (const float*)d_in[1];
    const float* mask   = (const float*)d_in[2];
    const float* Wux    = (const float*)d_in[3];
    const float* bu     = (const float*)d_in[4];
    const float* Wuh    = (const float*)d_in[5];
    const float* Wrx    = (const float*)d_in[6];
    const float* br     = (const float*)d_in[7];
    const float* Wrh    = (const float*)d_in[8];
    const float* Wcx    = (const float*)d_in[9];
    const float* bc     = (const float*)d_in[10];
    const float* Wch    = (const float*)d_in[11];
    float* out = (float*)d_out;

    (void)in_sizes; (void)n_in; (void)out_size; (void)d_ws; (void)ws_size;

    // ~231 KB dynamic LDS per workgroup (CDNA5 allows up to 320 KB/WGP).
    hipFuncSetAttribute((const void*)gru_scan_kernel,
                        hipFuncAttributeMaxDynamicSharedMemorySize,
                        (int)SMEM_BYTES);

    dim3 grid(B_TOT / BT);   // 128 workgroups, each owns 32 batch rows
    dim3 block(256);         // 8 wave32 waves
    gru_scan_kernel<<<grid, block, SMEM_BYTES, stream>>>(
        inputs, state, mask, Wux, bu, Wuh, Wrx, br, Wrh, Wcx, bc, Wch, out);
}
